// CustomLoss_46505905881568
// MI455X (gfx1250) — matrix operand order repaired and checked by reading, not compiled, not run on hardware
//
#include <hip/hip_runtime.h>
#include <math.h>

// ---------------- problem constants (match reference) ----------------
#define B_SZ   8192
#define D1_SZ  512
#define D_SZ   256
#define NNEG   100

// k_supp tiling: 64 row-blocks of 128 rows (8 waves x 16), 4 column splits,
// column panels of 64 cols x 256 K bf16 = 32 KB, double buffered in LDS.
#define PANEL_COLS    64
#define PANEL_SHORTS  (PANEL_COLS * D_SZ)      // 16384 shorts = 32 KB
#define PANEL_BYTES   (PANEL_SHORTS * 2)
#define PANELS_PER_CS 32                       // 8192 / 64 / 4 splits

// ---------------- vector types ----------------
typedef __bf16        v16bf __attribute__((ext_vector_type(16)));
typedef float         v8f   __attribute__((ext_vector_type(8)));
typedef float         v4f   __attribute__((ext_vector_type(4)));
typedef unsigned int  v4u   __attribute__((ext_vector_type(4)));
typedef int           v4i   __attribute__((ext_vector_type(4)));

// fp32 -> bf16, round-to-nearest-even
__device__ __forceinline__ unsigned short f2bf(float f) {
    unsigned int u = __float_as_uint(f);
    u += 0x7FFFu + ((u >> 16) & 1u);
    return (unsigned short)(u >> 16);
}

// A-matrix 16x32 bf16 fragment (ISA 7.12.2): lane l (row = base + (l&15)):
//   hi = l>=16 ; v0..v3 = K {8*hi .. 8*hi+7} ; v4..v7 = K {16+8*hi .. 16+8*hi+7}
__device__ __forceinline__ v16bf load_fragA(const unsigned short* rowp, int kc, int hi) {
    union { v16bf v; v4u q[2]; } u;
    u.q[0] = *reinterpret_cast<const v4u*>(rowp + kc + 8 * hi);
    u.q[1] = *reinterpret_cast<const v4u*>(rowp + kc + 16 + 8 * hi);
    return u.v;
}

// B-matrix 32x16 bf16 fragment: lane l (col = base + (l&15)):
//   hi = l>=16 ; v0..v3 = K {8*hi .. 8*hi+7} ; v4..v7 = K {16+8*hi .. 16+8*hi+7}
__device__ __forceinline__ v16bf load_fragB(const unsigned short* colp, int kc, int hi) {
    union { v16bf v; v4u q[2]; } u;
    u.q[0] = *reinterpret_cast<const v4u*>(colp + kc + 8 * hi);
    u.q[1] = *reinterpret_cast<const v4u*>(colp + kc + 16 + 8 * hi);
    return u.v;
}

// ---------------- CDNA5 async global->LDS copy (ASYNCcnt path) ----------------
// builtin signature (from hipcc diagnostic): (int4 AS(1)*, int4 AS(3)*, Imm, Imm)
__device__ __forceinline__ void async_cp16(void* lptr, const void* gaddr) {
    v4i* gp = (v4i*)const_cast<void*>(gaddr);
    v4i* lp = (v4i*)lptr;
#if __has_builtin(__builtin_amdgcn_global_load_async_to_lds_b128)
    __builtin_amdgcn_global_load_async_to_lds_b128(
        (__attribute__((address_space(1))) v4i*)gp,
        (__attribute__((address_space(3))) v4i*)lp,
        0, 0);
#else
    // GLOBAL_LOAD_ASYNC_TO_LDS_B128: VDST = LDS byte offset VGPR, VADDR = 64-bit addr
    unsigned int lds_byte = (unsigned int)(unsigned long long)
                            (__attribute__((address_space(3))) char*)(char*)lptr;
    asm volatile("global_load_async_to_lds_b128 %0, %1, off"
                 :: "v"(lds_byte), "v"(gp)
                 : "memory");
#endif
}

__device__ __forceinline__ void wait_async0() {
#if __has_builtin(__builtin_amdgcn_s_wait_asynccnt)
    __builtin_amdgcn_s_wait_asynccnt(0);
#else
    asm volatile("s_wait_asynccnt 0x0" ::: "memory");
#endif
}

// ---------------- init: zero accumulators + per-row partials + output ----------------
__global__ void k_init(float* acc, float* rowred, float* out) {
    int gid    = blockIdx.x * blockDim.x + threadIdx.x;
    int stride = gridDim.x * blockDim.x;
    for (int i = gid; i < 3 * B_SZ; i += stride) rowred[i] = 0.0f;
    if (gid < 8) acc[gid] = 0.0f;
    if (gid == 0) out[0] = 0.0f;
}

// ---------------- MAE + ||W||^2 partial sums ----------------
__global__ void k_scalar_red(const float* __restrict__ yp, const float* __restrict__ yt,
                             const float* __restrict__ W, float* acc) {
    int gid    = blockIdx.x * blockDim.x + threadIdx.x;
    int stride = gridDim.x * blockDim.x;
    float sa = 0.0f, sw = 0.0f;
    for (int i = gid; i < D1_SZ * D_SZ; i += stride) {
        float w = W[i];
        sw += w * w;
        if (i < B_SZ) sa += fabsf(yp[i] - yt[i]);
    }
    for (int m = 16; m >= 1; m >>= 1) {
        sa += __shfl_xor(sa, m, 32);
        sw += __shfl_xor(sw, m, 32);
    }
    if ((threadIdx.x & 31) == 0) {
        atomicAdd(&acc[0], sa);
        atomicAdd(&acc[1], sw);
    }
}

// ---------------- pack e2 (f32 -> bf16) ----------------
__global__ void k_pack_e2(const float* __restrict__ e2, unsigned short* __restrict__ e2b) {
    int gid    = blockIdx.x * blockDim.x + threadIdx.x;
    int stride = gridDim.x * blockDim.x;
    for (int i = gid; i < B_SZ * D_SZ; i += stride) e2b[i] = f2bf(e2[i]);
}

// ---------------- pack W transposed (Wt[n,k] = W[k,n], bf16) ----------------
__global__ void k_pack_wt(const float* __restrict__ W, unsigned short* __restrict__ wtb) {
    int gid    = blockIdx.x * blockDim.x + threadIdx.x;
    int stride = gridDim.x * blockDim.x;
    for (int i = gid; i < D_SZ * D1_SZ; i += stride) {
        int n = i >> 9;          // / 512
        int k = i & (D1_SZ - 1); // % 512
        wtb[i] = f2bf(W[k * D_SZ + n]);
    }
}

// ---------------- mixup + pack: aug_e1 = l*e1[b] + (1-l)*e1[a], bf16 ----------------
__global__ void k_mixup(const float* __restrict__ e1, const float* __restrict__ lu,
                        const int* __restrict__ aidx, unsigned short* __restrict__ augb) {
    int gid    = blockIdx.x * blockDim.x + threadIdx.x;
    int stride = gridDim.x * blockDim.x;
    for (int i = gid; i < B_SZ * D1_SZ; i += stride) {
        int b = i >> 9;          // / 512
        int k = i & (D1_SZ - 1);
        float l = 0.9f + 0.1f * lu[b];
        int a = aidx[b];
        a += (a >= b) ? 1 : 0;
        float v = l * e1[i] + (1.0f - l) * e1[a * D1_SZ + k];
        augb[i] = f2bf(v);
    }
}

// ---------------- phi = aug_e1(bf16) @ W(bf16)  [8192x512]x[512x256] -> f32 ----------------
// one 16x16 output tile per wave; K=512 -> 16 WMMA per tile; 1024 blocks x 8 waves = 8192 tiles
__global__ void k_phi_gemm(const unsigned short* __restrict__ augb,
                           const unsigned short* __restrict__ wtb,
                           float* __restrict__ phi) {
    int tid  = threadIdx.x;
    int wave = tid >> 5;
    int lane = tid & 31;
    int l15  = lane & 15;
    int hi   = lane >> 4;

    int tile    = blockIdx.x * 8 + wave;   // 8192 tiles
    int mt      = tile >> 4;               // 512 row tiles
    int nt      = tile & 15;               // 16 col tiles
    int rowbase = mt * 16;
    int colbase = nt * 16;

    const unsigned short* arow = augb + (size_t)(rowbase + l15) * D1_SZ;
    const unsigned short* brow = wtb  + (size_t)(colbase + l15) * D1_SZ;

    v8f c = {};
#pragma unroll
    for (int kc = 0; kc < D1_SZ; kc += 32) {
        v16bf a = load_fragA(arow, kc, hi);
        v16bf b = load_fragB(brow, kc, hi);
        c = __builtin_amdgcn_wmma_f32_16x16x32_bf16(false, a, false, b, (short)0, c,
                                                    false, false);
    }
#pragma unroll
    for (int v = 0; v < 8; ++v) {
        int r = rowbase + v + 8 * hi;
        phi[(size_t)r * D_SZ + colbase + l15] = c[v];
    }
}

// ---------------- L_supp main: masked row sums of exp(1e-10 * e2@e2^T) ----------------
// grid = 64 row-blocks x 4 col-splits. Block: 8 waves, each owns 16 rows (A fragments
// pinned in 64 VGPRs). Column panels (64 cols x 256 K bf16 = 32 KB) are async-copied
// global->LDS (double buffered, ASYNCcnt) and shared by all 8 waves: 128 FLOP per L2
// byte instead of 16. Per-row partials -> global atomics; log taken in k_supp_final.
__global__ void k_supp(const unsigned short* __restrict__ e2b,
                       const int* __restrict__ tag,
                       float* __restrict__ nomR, float* __restrict__ denR,
                       float* __restrict__ cntR) {
    extern __shared__ unsigned short panels[];   // 2 * PANEL_SHORTS (64 KB dynamic)

    const int tid  = threadIdx.x;
    const int wave = tid >> 5;
    const int lane = tid & 31;
    const int l15  = lane & 15;
    const int hi   = lane >> 4;

    const int rb      = blockIdx.x >> 2;                 // 0..63
    const int cs      = blockIdx.x & 3;                  // 0..3
    const int rowbase = rb * 128 + wave * 16;
    const int pan0    = cs * PANELS_PER_CS;              // first 64-col panel index

    // A fragments for this wave's 16 rows, K=256 (8 x v16bf in registers)
    const unsigned short* arow = e2b + (size_t)(rowbase + l15) * D_SZ;
    v16bf A[8];
#pragma unroll
    for (int kc = 0; kc < 8; ++kc) A[kc] = load_fragA(arow, kc * 32, hi);

    int trow[8];
#pragma unroll
    for (int v = 0; v < 8; ++v) trow[v] = tag[rowbase + v + 8 * hi];

    float nom[8], den[8], cnt[8];
#pragma unroll
    for (int v = 0; v < 8; ++v) { nom[v] = 0.0f; den[v] = 0.0f; cnt[v] = 0.0f; }

    // async-stage one 32 KB panel: 2048 16B chunks, 8 per thread
    auto stage = [&](int p) {
        const unsigned short* g = e2b + (size_t)(pan0 + p) * PANEL_SHORTS;
        unsigned short* lbase = panels + (p & 1) * PANEL_SHORTS;
#pragma unroll
        for (int j = 0; j < 8; ++j) {
            int chunk = tid + j * 256;
            async_cp16(lbase + chunk * 8, g + chunk * 8);
        }
    };

    stage(0);
    for (int p = 0; p < PANELS_PER_CS; ++p) {
        wait_async0();        // my loads for panel p done
        __syncthreads();      // everyone's loads for panel p done
        if (p + 1 < PANELS_PER_CS) stage(p + 1);   // prefetch into other buffer

        const unsigned short* pb = panels + (p & 1) * PANEL_SHORTS;
        const int colpan = (pan0 + p) * PANEL_COLS;
#pragma unroll
        for (int c = 0; c < 4; ++c) {              // 4 16-col tiles per panel
            const unsigned short* brow = pb + (c * 16 + l15) * D_SZ;
            v8f cc = {};
#pragma unroll
            for (int kc = 0; kc < 8; ++kc) {
                v16bf bfr = load_fragB(brow, kc * 32, hi);
                cc = __builtin_amdgcn_wmma_f32_16x16x32_bf16(false, A[kc], false, bfr,
                                                             (short)0, cc, false, false);
            }
            int tcol = tag[colpan + c * 16 + l15];
#pragma unroll
            for (int v = 0; v < 8; ++v) {
                float s = __expf(1e-10f * cc[v]);
                float d = (tcol != trow[v]) ? 1.0f : 0.0f;
                nom[v] += d * s;
                den[v] += s;
                cnt[v] += d;
            }
        }
        __syncthreads();      // all waves done reading buffer p before it is re-staged
    }

    // reduce over the 16 columns held by each lane-half
#pragma unroll
    for (int m = 8; m >= 1; m >>= 1) {
#pragma unroll
        for (int v = 0; v < 8; ++v) {
            nom[v] += __shfl_xor(nom[v], m, 32);
            den[v] += __shfl_xor(den[v], m, 32);
            cnt[v] += __shfl_xor(cnt[v], m, 32);
        }
    }
    if (l15 == 0) {           // lanes 0 / 16 carry rows {v} / {v+8}
#pragma unroll
        for (int v = 0; v < 8; ++v) {
            int r = rowbase + v + 8 * hi;
            atomicAdd(&nomR[r], nom[v]);
            atomicAdd(&denR[r], den[v]);
            atomicAdd(&cntR[r], cnt[v]);
        }
    }
}

// ---------------- L_supp epilogue: per-row log + global sum ----------------
__global__ void k_supp_final(const float* __restrict__ nomR, const float* __restrict__ denR,
                             const float* __restrict__ cntR, float* acc) {
    int gid = blockIdx.x * blockDim.x + threadIdx.x;
    float L = 0.0f;
    if (gid < B_SZ && cntR[gid] > 0.0f) L = __logf(nomR[gid] / (denR[gid] + 1e-6f));
#pragma unroll
    for (int m = 16; m >= 1; m >>= 1) L += __shfl_xor(L, m, 32);
    if ((threadIdx.x & 31) == 0) atomicAdd(&acc[3], L);
}

// ---------------- L_aug: gathered InfoNCE rows over phi_aug (L2-resident) ----------------
__global__ void k_aug(const float* __restrict__ phi, const float* __restrict__ e2,
                      const int* __restrict__ neg_idx, float* acc) {
    __shared__ __align__(16) float e2s[D_SZ];
    __shared__ float sums[2];   // [0]=pos, [1]=sum(neg)

    int tid = threadIdx.x;
    int b   = blockIdx.x;
    e2s[tid] = e2[(size_t)b * D_SZ + tid];
    if (tid < 2) sums[tid] = 0.0f;
    __syncthreads();

    int wave = tid >> 5;
    int lane = tid & 31;
    const v4f* e4 = reinterpret_cast<const v4f*>(e2s);
    v4f ea = e4[lane * 2];
    v4f eb = e4[lane * 2 + 1];

    for (int item = wave; item < NNEG + 1; item += 8) {
        int row;
        if (item == 0) {
            row = b;
        } else {
            int ni = neg_idx[(size_t)b * NNEG + (item - 1)];
            row = ni + ((ni >= b) ? 1 : 0);
        }
        const v4f* pr = reinterpret_cast<const v4f*>(phi + (size_t)row * D_SZ);
        v4f m = pr[lane * 2] * ea + pr[lane * 2 + 1] * eb;
        float s = m.x + m.y + m.z + m.w;
#pragma unroll
        for (int msk = 16; msk >= 1; msk >>= 1) s += __shfl_xor(s, msk, 32);
        if (lane == 0) atomicAdd(&sums[item == 0 ? 0 : 1], __expf(1e-6f * s));
    }
    __syncthreads();

    if (tid == 0) {
        float pos = sums[0];
        float L = __logf(pos / (1e-6f + pos + sums[1]));
        atomicAdd(&acc[2], L);
    }
}

// ---------------- combine ----------------
__global__ void k_finalize(const float* acc, float* out) {
    if (threadIdx.x == 0) {
        float mse  = acc[0] / (float)B_SZ;
        float reg  = 1e-4f * sqrtf(acc[1]);
        float aug  = 0.1f  * (-(acc[2] / (float)B_SZ));
        float supp = 1e-3f * (-(acc[3] / (float)B_SZ));
        out[0] = mse + reg + aug + supp;
    }
}

// ---------------- workspace layout (bytes) ----------------
//   [0, 256)        : 8-float accumulator block
//   WS_E2B  +4MiB   : e2 bf16        8192*256*2
//   WS_AUGB +8MiB   : aug_e1 bf16    8192*512*2
//   WS_WTB  +256KiB : W^T bf16       256*512*2
//   WS_PHI  +8MiB   : phi f32        8192*256*4
//   WS_ROW  +96KiB  : per-row nom/den/cnt partials (3 * 8192 f32)
// total ~20.3 MiB
#define WS_E2B   256
#define WS_AUGB  (WS_E2B  + B_SZ * D_SZ * 2)
#define WS_WTB   (WS_AUGB + B_SZ * D1_SZ * 2)
#define WS_PHI   (WS_WTB  + D_SZ * D1_SZ * 2)
#define WS_ROW   (WS_PHI  + B_SZ * D_SZ * 4)

extern "C" void kernel_launch(void* const* d_in, const int* in_sizes, int n_in,
                              void* d_out, int out_size, void* d_ws, size_t ws_size,
                              hipStream_t stream) {
    (void)in_sizes; (void)n_in; (void)out_size; (void)ws_size;

    const float* e1        = (const float*)d_in[0];
    const float* e2        = (const float*)d_in[1];
    const float* y_pred    = (const float*)d_in[2];
    const float* y_target  = (const float*)d_in[3];
    const float* W_e2      = (const float*)d_in[4];
    const float* lmbda_u   = (const float*)d_in[5];
    const int*   dom_tag   = (const int*)d_in[6];
    const int*   aug_nidx  = (const int*)d_in[7];
    const int*   neg_idx   = (const int*)d_in[8];
    float*       out       = (float*)d_out;

    char* ws = (char*)d_ws;
    float*          acc    = (float*)ws;
    unsigned short* e2b    = (unsigned short*)(ws + WS_E2B);
    unsigned short* augb   = (unsigned short*)(ws + WS_AUGB);
    unsigned short* wtb    = (unsigned short*)(ws + WS_WTB);
    float*          phi    = (float*)(ws + WS_PHI);
    float*          rowred = (float*)(ws + WS_ROW);
    float*          nomR   = rowred;
    float*          denR   = rowred + B_SZ;
    float*          cntR   = rowred + 2 * B_SZ;

    k_init      <<<96,  256, 0, stream>>>(acc, rowred, out);
    k_scalar_red<<<128, 256, 0, stream>>>(y_pred, y_target, W_e2, acc);
    k_pack_e2   <<<1024,256, 0, stream>>>(e2, e2b);
    k_pack_wt   <<<512, 256, 0, stream>>>(W_e2, wtb);
    k_mixup     <<<2048,256, 0, stream>>>(e1, lmbda_u, aug_nidx, augb);
    k_phi_gemm  <<<1024,256, 0, stream>>>(augb, wtb, phi);
    k_supp      <<<256, 256, 2 * PANEL_BYTES, stream>>>(e2b, dom_tag, nomR, denR, cntR);
    k_supp_final<<<32,  256, 0, stream>>>(nomR, denR, cntR, acc);
    k_aug       <<<8192,256, 0, stream>>>(phi, e2, neg_idx, acc);
    k_finalize  <<<1,   32,  0, stream>>>(acc, out);
}
